// TFData2VecVisionSelfAttention_11355893530990
// MI455X (gfx1250) — matrix operand order repaired and checked by reading, not compiled, not run on hardware
//
#include <hip/hip_runtime.h>
#include <hip/hip_bf16.h>

// ---------------------------------------------------------------------------
// TFData2VecVisionSelfAttention for MI455X (gfx1250, wave32, WMMA).
// Split-bf16 (hi+lo) WMMA for ~f32 accuracy at bf16-WMMA rate.
// v2: splits are computed once and stored packed (hi|lo u32); fragment builds
//     are pure bit-merges; LDS staging vectorized to b128.
// v2.1: merge_pair returns by value (vector elements can't bind to refs).
// ---------------------------------------------------------------------------

typedef __attribute__((ext_vector_type(16))) __bf16        v16bf;
typedef __attribute__((ext_vector_type(8)))  float         v8f;
typedef __attribute__((ext_vector_type(8)))  unsigned int  v8u;

#define N_B   16
#define N_S   577
#define N_D   768
#define N_H   12
#define HEADD 64
#define BSROWS (N_B * N_S)          // 9232
#define SPAD  640                    // padded seq (5 q-tiles of 128; 19 kv steps of 32 = 608 <= 640)
#define BIAS_KPAD 592                // padded k width for the dense bias array
#define QKV_ELEMS ((size_t)N_B * N_H * SPAD * HEADD)   // 7,864,320 elems per tensor
#define NEG_BIG (-3.0e38f)

struct bf2 { __bf16 hi, lo; };
struct u2  { unsigned int hi, lo; };

static __device__ __forceinline__ bf2 split2(float x) {
  bf2 r;
  r.hi = (__bf16)x;                       // RNE
  r.lo = (__bf16)(x - (float)r.hi);       // residual
  return r;
}

// packed split word: bits[15:0] = hi bf16, bits[31:16] = lo bf16
static __device__ __forceinline__ unsigned int pack_split(float x) {
  bf2 r = split2(x);
  unsigned short uh, ul;
  __builtin_memcpy(&uh, &r.hi, 2);
  __builtin_memcpy(&ul, &r.lo, 2);
  return (unsigned int)uh | ((unsigned int)ul << 16);
}

static __device__ __forceinline__ v16bf asbf(v8u v) {
  return __builtin_bit_cast(v16bf, v);
}

// merge two packed words (even element w0, odd element w1) into one VGPR each
// of the hi-fragment and lo-fragment.
static __device__ __forceinline__ u2 merge_pair(unsigned int w0, unsigned int w1) {
  u2 r;
  r.hi = (w0 & 0xffffu) | (w1 << 16);
  r.lo = (w0 >> 16) | (w1 & 0xffff0000u);
  return r;
}

static __device__ __forceinline__ v8f wmma_bf(v16bf a, v16bf b, v8f c) {
  // D = A(16x32 bf16) * B(32x16 bf16) + C(16x16 f32)
  return __builtin_amdgcn_wmma_f32_16x16x32_bf16(false, a, false, b, (short)0, c,
                                                 false, false);
}

// ---------------------------------------------------------------------------
// Kernel 1: dense relative-position bias  Bias[h][q][k] = table[idx[q,k], h]
// ---------------------------------------------------------------------------
__global__ void bias_gather_kernel(const float* __restrict__ table,
                                   const int* __restrict__ relidx,
                                   float* __restrict__ bias) {
  int idx = blockIdx.x * 256 + threadIdx.x;
  const int total = N_H * N_S * BIAS_KPAD;
  if (idx >= total) return;
  int k = idx % BIAS_KPAD;
  int t = idx / BIAS_KPAD;
  int q = t % N_S;
  int h = t / N_S;
  float v = 0.0f;
  if (k < N_S) v = table[relidx[q * N_S + k] * N_H + h];
  bias[idx] = v;
}

// ---------------------------------------------------------------------------
// Kernel 2: fused QKV projection. blockIdx.z selects {Q,K,V}.
// Q -> f32 [B,H,SPAD,64], pre-scaled by 1/8.  K,V -> packed-split u32, same layout.
// Block tile 128(M) x 64(N); 8 waves; LDS tiles staged pre-split.
// ---------------------------------------------------------------------------
__global__ __launch_bounds__(256) void qkv_gemm_kernel(
    const float* __restrict__ X, const float* __restrict__ Wq,
    const float* __restrict__ bq, const float* __restrict__ Wk,
    const float* __restrict__ Wv, const float* __restrict__ bv,
    float* __restrict__ ws) {
  const int z = blockIdx.z;
  const float* W = (z == 0) ? Wq : ((z == 1) ? Wk : Wv);
  const float* bias = (z == 0) ? bq : ((z == 2) ? bv : nullptr);
  float* OutF = ws;                                        // Q region (f32)
  unsigned int* OutU = (unsigned int*)(ws + (size_t)z * QKV_ELEMS);  // K/V region
  const float scale = (z == 0) ? 0.125f : 1.0f;

  const int tileM = blockIdx.x * 128;
  const int tileN = blockIdx.y * 64;

  __shared__ __attribute__((aligned(16))) unsigned int ldsA[128 * 36]; // packed split
  __shared__ __attribute__((aligned(16))) unsigned int ldsB[32 * 64];  // packed split

  const int tid = threadIdx.x;
  const int wave = tid >> 5;
  const int lane = tid & 31;
  const int l16 = lane & 15;
  const int lhalf = lane >> 4;

  v8f zero8 = {};
  v8f acc[4] = {zero8, zero8, zero8, zero8};

  for (int kt = 0; kt < 24; ++kt) {
    const int k0 = kt * 32;
    __syncthreads();
    // stage A tile 128x32 as packed split (split done once per element)
    for (int i = tid; i < 1024; i += 256) {          // 1024 float4
      int row = i >> 3;
      int c4 = (i & 7) << 2;
      int m = tileM + row;
      float4 x = (m < BSROWS)
                     ? *(const float4*)(X + (size_t)m * N_D + k0 + c4)
                     : make_float4(0.f, 0.f, 0.f, 0.f);
      uint4 p;
      p.x = pack_split(x.x); p.y = pack_split(x.y);
      p.z = pack_split(x.z); p.w = pack_split(x.w);
      *(uint4*)&ldsA[row * 36 + c4] = p;
    }
    // stage B tile 32x64 as packed split
    for (int i = tid; i < 512; i += 256) {           // 512 float4
      int row = i >> 4;
      int c4 = (i & 15) << 2;
      float4 x = *(const float4*)(W + (size_t)(k0 + row) * N_D + tileN + c4);
      uint4 p;
      p.x = pack_split(x.x); p.y = pack_split(x.y);
      p.z = pack_split(x.z); p.w = pack_split(x.w);
      *(uint4*)&ldsB[row * 64 + c4] = p;
    }
    __syncthreads();

    // A fragment (bit-merge only)
    v8u a_hib, a_lob;
    const unsigned int* arowp = ldsA + (wave * 16 + l16) * 36;
#pragma unroll
    for (int jj = 0; jj < 8; ++jj) {
      int kl = 2 * jj + (lhalf << 4);
      u2 m = merge_pair(arowp[kl], arowp[kl + 1]);
      a_hib[jj] = m.hi;
      a_lob[jj] = m.lo;
    }
    const v16bf a_hi = asbf(a_hib), a_lo = asbf(a_lob);

#pragma unroll
    for (int nn = 0; nn < 4; ++nn) {
      v8u b_hib, b_lob;
      const int ncol = nn * 16 + l16;
#pragma unroll
      for (int jj = 0; jj < 8; ++jj) {
        int kl = 2 * jj + (lhalf << 4);
        u2 m = merge_pair(ldsB[kl * 64 + ncol], ldsB[(kl + 1) * 64 + ncol]);
        b_hib[jj] = m.hi;
        b_lob[jj] = m.lo;
      }
      const v16bf b_hi = asbf(b_hib), b_lo = asbf(b_lob);
      acc[nn] = wmma_bf(a_hi, b_hi, acc[nn]);
      acc[nn] = wmma_bf(a_lo, b_hi, acc[nn]);
      acc[nn] = wmma_bf(a_hi, b_lo, acc[nn]);
    }
  }

  // epilogue: +bias, scale, scatter to [B,H,SPAD,64] (Q as f32, K/V packed)
#pragma unroll
  for (int nn = 0; nn < 4; ++nn) {
    const int n = tileN + nn * 16 + l16;
    const float badd = bias ? bias[n] : 0.0f;
    const int h = n >> 6, d = n & 63;
#pragma unroll
    for (int r = 0; r < 8; ++r) {
      int m = tileM + wave * 16 + r + (lhalf << 3);
      if (m < BSROWS) {
        int bb = m / N_S, s = m % N_S;
        float v = (acc[nn][r] + badd) * scale;
        size_t off = (((size_t)(bb * N_H + h)) * SPAD + s) * HEADD + d;
        if (z == 0) OutF[off] = v;
        else        OutU[off] = pack_split(v);
      }
    }
  }
}

// ---------------------------------------------------------------------------
// Kernel 3: flash attention. grid = (5 q-tiles, 192 bh). Block = 256 (8 waves).
// K/V arrive pre-split-packed; LDS staging is a straight b128 copy.
// ---------------------------------------------------------------------------
__global__ __launch_bounds__(256) void attn_kernel(
    const float* __restrict__ Q, const unsigned int* __restrict__ K,
    const unsigned int* __restrict__ V, const float* __restrict__ Bias,
    float* __restrict__ out) {
  const int bh = blockIdx.y;
  const int b = bh / N_H, h = bh % N_H;
  const int qtile = blockIdx.x * 128;

  __shared__ __attribute__((aligned(16))) unsigned int ldsK[32 * 68];
  __shared__ __attribute__((aligned(16))) unsigned int ldsV[32 * 68];
  __shared__ unsigned int ldsP[8 * 16 * 33];   // per-wave P tile, packed split

  const int tid = threadIdx.x;
  const int wave = tid >> 5;
  const int lane = tid & 31;
  const int l16 = lane & 15;
  const int lhalf = lane >> 4;

  const size_t stride = (size_t)SPAD * HEADD;
  const float* Qb = Q + (size_t)bh * stride;
  const unsigned int* Kb = K + (size_t)bh * stride;
  const unsigned int* Vb = V + (size_t)bh * stride;
  const float* biasH = Bias + (size_t)h * N_S * BIAS_KPAD;

  // Q fragments (split once per block)
  v16bf qa_hi[2], qa_lo[2];
  {
    const int qrow = qtile + wave * 16 + l16;    // < 640 always
    const float* qp = Qb + (size_t)qrow * HEADD;
#pragma unroll
    for (int c = 0; c < 2; ++c) {
#pragma unroll
      for (int e = 0; e < 16; ++e) {
        int dim = c * 32 + e + (lhalf << 4);
        bf2 t = split2(qp[dim]);
        qa_hi[c][e] = t.hi;
        qa_lo[c][e] = t.lo;
      }
    }
  }

  v8f zero8 = {};
  v8f oacc[4] = {zero8, zero8, zero8, zero8};
  float m_run[8], l_run[8];
#pragma unroll
  for (int r = 0; r < 8; ++r) { m_run[r] = NEG_BIG; l_run[r] = 0.0f; }

  const int nsteps = 19;  // 19*32 = 608 >= 577
  for (int step = 0; step < nsteps; ++step) {
    const int kv0 = step * 32;
    __syncthreads();
    for (int i = tid; i < 512; i += 256) {       // 512 uint4 per tensor tile
      int row = i >> 4;
      int c4 = (i & 15) << 2;
      *(uint4*)&ldsK[row * 68 + c4] =
          *(const uint4*)(Kb + (size_t)(kv0 + row) * HEADD + c4);
      *(uint4*)&ldsV[row * 68 + c4] =
          *(const uint4*)(Vb + (size_t)(kv0 + row) * HEADD + c4);
    }
    if (step + 1 < nsteps) {   // hint next tiles toward the caches
      __builtin_prefetch(Kb + (size_t)(kv0 + 32) * HEADD + lane * 8, 0, 0);
      __builtin_prefetch(Vb + (size_t)(kv0 + 32) * HEADD + lane * 8, 0, 0);
    }
    __syncthreads();

    // ---- scores: S[16q x 32k] = Q(16x64) . K^T ----
    v8f sc0 = zero8, sc1 = zero8;
#pragma unroll
    for (int c = 0; c < 2; ++c) {
      v8u kb_hib, kb_lob;
      // j = 0: kv rows kv0 + [0,16)
      const unsigned int* kr0 = ldsK + l16 * 68;
#pragma unroll
      for (int jj = 0; jj < 8; ++jj) {
        int dim = c * 32 + 2 * jj + (lhalf << 4);
        u2 m = merge_pair(kr0[dim], kr0[dim + 1]);
        kb_hib[jj] = m.hi;
        kb_lob[jj] = m.lo;
      }
      {
        const v16bf kb_hi = asbf(kb_hib), kb_lo = asbf(kb_lob);
        sc0 = wmma_bf(qa_hi[c], kb_hi, sc0);
        sc0 = wmma_bf(qa_lo[c], kb_hi, sc0);
        sc0 = wmma_bf(qa_hi[c], kb_lo, sc0);
      }
      // j = 1: kv rows kv0 + [16,32)
      const unsigned int* kr1 = ldsK + (16 + l16) * 68;
#pragma unroll
      for (int jj = 0; jj < 8; ++jj) {
        int dim = c * 32 + 2 * jj + (lhalf << 4);
        u2 m = merge_pair(kr1[dim], kr1[dim + 1]);
        kb_hib[jj] = m.hi;
        kb_lob[jj] = m.lo;
      }
      {
        const v16bf kb_hi = asbf(kb_hib), kb_lo = asbf(kb_lob);
        sc1 = wmma_bf(qa_hi[c], kb_hi, sc1);
        sc1 = wmma_bf(qa_lo[c], kb_hi, sc1);
        sc1 = wmma_bf(qa_hi[c], kb_lo, sc1);
      }
    }

    // ---- bias + mask + online softmax (per row r) ----
    const int k0g = kv0 + l16;
    const int k1g = kv0 + 16 + l16;
    unsigned int* pRow = ldsP + wave * (16 * 33);
#pragma unroll
    for (int r = 0; r < 8; ++r) {
      const int qrow = qtile + wave * 16 + r + (lhalf << 3);
      float s0 = sc0[r], s1 = sc1[r];
      const bool qok = qrow < N_S;
      if (qok && k0g < N_S) s0 += biasH[(size_t)qrow * BIAS_KPAD + k0g];
      if (k0g >= N_S) s0 = NEG_BIG;
      if (qok && k1g < N_S) s1 += biasH[(size_t)qrow * BIAS_KPAD + k1g];
      if (k1g >= N_S) s1 = NEG_BIG;

      // row reduction across the 16 lanes of this half (xor masks stay in-half)
      float t = fmaxf(s0, s1);
      t = fmaxf(t, __shfl_xor(t, 1));
      t = fmaxf(t, __shfl_xor(t, 2));
      t = fmaxf(t, __shfl_xor(t, 4));
      t = fmaxf(t, __shfl_xor(t, 8));
      const float mnew = fmaxf(m_run[r], t);
      const float alpha = __expf(m_run[r] - mnew);
      m_run[r] = mnew;
      const float p0 = __expf(s0 - mnew);
      const float p1 = __expf(s1 - mnew);
      float rs = p0 + p1;
      rs += __shfl_xor(rs, 1);
      rs += __shfl_xor(rs, 2);
      rs += __shfl_xor(rs, 4);
      rs += __shfl_xor(rs, 8);
      l_run[r] = l_run[r] * alpha + rs;
#pragma unroll
      for (int nn = 0; nn < 4; ++nn) oacc[nn][r] *= alpha;
      const int prow = r + (lhalf << 3);
      pRow[prow * 33 + l16]      = pack_split(p0);
      pRow[prow * 33 + 16 + l16] = pack_split(p1);
    }

    // ---- P(16x32) -> A fragment (in-wave LDS transpose; DS ops in-order) ----
    v8u pa_hib, pa_lob;
    const unsigned int* prd = pRow + l16 * 33;
#pragma unroll
    for (int jj = 0; jj < 8; ++jj) {
      int kl = 2 * jj + (lhalf << 4);
      u2 m = merge_pair(prd[kl], prd[kl + 1]);
      pa_hib[jj] = m.hi;
      pa_lob[jj] = m.lo;
    }
    const v16bf pa_hi = asbf(pa_hib), pa_lo = asbf(pa_lob);

    // ---- O += P . V ----
#pragma unroll
    for (int nn = 0; nn < 4; ++nn) {
      v8u vb_hib, vb_lob;
      const int dcol = nn * 16 + l16;
#pragma unroll
      for (int jj = 0; jj < 8; ++jj) {
        int kl = 2 * jj + (lhalf << 4);
        u2 m = merge_pair(ldsV[kl * 68 + dcol], ldsV[(kl + 1) * 68 + dcol]);
        vb_hib[jj] = m.hi;
        vb_lob[jj] = m.lo;
      }
      const v16bf vb_hi = asbf(vb_hib), vb_lo = asbf(vb_lob);
      oacc[nn] = wmma_bf(pa_hi, vb_hi, oacc[nn]);
      oacc[nn] = wmma_bf(pa_lo, vb_hi, oacc[nn]);
      oacc[nn] = wmma_bf(pa_hi, vb_lo, oacc[nn]);
    }
  }

  // ---- normalize + store: out[b, s, h*64 + d] ----
#pragma unroll
  for (int r = 0; r < 8; ++r) {
    const int qrow = qtile + wave * 16 + r + (lhalf << 3);
    if (qrow < N_S) {
      const float inv = 1.0f / l_run[r];
      const size_t base = ((size_t)b * N_S + qrow) * N_D + h * HEADD;
#pragma unroll
      for (int nn = 0; nn < 4; ++nn)
        out[base + nn * 16 + l16] = oacc[nn][r] * inv;
    }
  }
}

// ---------------------------------------------------------------------------
extern "C" void kernel_launch(void* const* d_in, const int* in_sizes, int n_in,
                              void* d_out, int out_size, void* d_ws, size_t ws_size,
                              hipStream_t stream) {
  const float* X     = (const float*)d_in[0];
  const float* Wq    = (const float*)d_in[1];
  const float* bq    = (const float*)d_in[2];
  const float* Wk    = (const float*)d_in[3];
  const float* Wv    = (const float*)d_in[4];
  const float* bv    = (const float*)d_in[5];
  const float* table = (const float*)d_in[6];
  const int*   ridx  = (const int*)d_in[7];

  float* ws = (float*)d_ws;                         // ~111 MB total:
  float*        Qw   = ws;                          //  Q  [16,12,640,64] f32
  unsigned int* Kw   = (unsigned int*)(ws + QKV_ELEMS);      // K packed split u32
  unsigned int* Vw   = (unsigned int*)(ws + 2 * QKV_ELEMS);  // V packed split u32
  float*        Bias = ws + 3 * QKV_ELEMS;          //  bias [12,577,592] f32

  (void)in_sizes; (void)n_in; (void)out_size; (void)ws_size;

  const int biasTotal = N_H * N_S * BIAS_KPAD;
  bias_gather_kernel<<<(biasTotal + 255) / 256, 256, 0, stream>>>(table, ridx, Bias);

  // M tiles: ceil(9232/128)=73; N tiles: 768/64=12; z: Q/K/V
  qkv_gemm_kernel<<<dim3(73, 12, 3), 256, 0, stream>>>(X, Wq, bq, Wk, Wv, bv, ws);

  // q tiles: ceil(577/128)=5; bh = 16*12 = 192
  attn_kernel<<<dim3(5, 192), 256, 0, stream>>>(Qw, Kw, Vw, Bias, (float*)d_out);
}
